// Nucleus_91087666413604
// MI455X (gfx1250) — compile-verified, matrix operand order checked by reading, not run on hardware
//
#include <hip/hip_runtime.h>
#include <hip/hip_bf16.h>

// ---------------- problem constants ----------------
#define BB    8
#define SS    256
#define DD    1024
#define VV    50257
#define NHH   8
#define DH    128
#define HIDD  4096
#define LLYR  2
#define NACTT 2000
#define TOPKK 20
#define NTOK  (BB * SS)        // 2048
#define NCE   (BB * (SS - 1))  // 2040
#define NTILE_V ((VV + 15) / 16)  // 3142
#define CE_CHUNKS 8

// ---------------- WMMA plumbing ----------------
typedef float  v8f   __attribute__((ext_vector_type(8)));
typedef float  v16f  __attribute__((ext_vector_type(16)));
typedef __bf16 v16bf __attribute__((ext_vector_type(16)));

union BFrag {
    v16bf v;
    unsigned short u[16];
    uint4 q[2];
};

// hardware packed conversion path (v_cvt_pk_bf16_f32), RNE rounding
__device__ __forceinline__ unsigned short f2bf(float f) {
    __bf16 b = (__bf16)f;
    return __builtin_bit_cast(unsigned short, b);
}

__device__ __forceinline__ v8f wmma_bf16(const BFrag& a, const BFrag& b, v8f c) {
    return __builtin_amdgcn_wmma_f32_16x16x32_bf16(false, a.v, false, b.v,
                                                   (short)0, c, false, false);
}

__device__ __forceinline__ void pack16(BFrag& pk, float4 f0, float4 f1, float4 f2, float4 f3) {
    v16f t;
    t[0] = f0.x;  t[1] = f0.y;  t[2] = f0.z;  t[3] = f0.w;
    t[4] = f1.x;  t[5] = f1.y;  t[6] = f1.z;  t[7] = f1.w;
    t[8] = f2.x;  t[9] = f2.y;  t[10] = f2.z; t[11] = f2.w;
    t[12] = f3.x; t[13] = f3.y; t[14] = f3.z; t[15] = f3.w;
    pk.v = __builtin_convertvector(t, v16bf);
}

// ---------------- GEMM: Y = X @ W^T + bias (opt. relu) ----------------
// X: M x K (row stride ldx), W: N x K (row-major), Y: M x N (row stride ldy)
#define GBM 128
#define GBN 128
#define GBK 32
#define LSTR 48   // ushort row stride in LDS (16B aligned, conflict padding)

__global__ __launch_bounds__(256)
void gemm_xwT(const float* __restrict__ X, long ldx,
              const float* __restrict__ W,
              const float* __restrict__ bias,
              float* __restrict__ Y, long ldy,
              int M, int N, int K, int relu)
{
    __shared__ unsigned short As[GBM * LSTR];
    __shared__ unsigned short Bs[GBN * LSTR];

    int tid = threadIdx.x, lane = tid & 31, w = tid >> 5;
    int bm = blockIdx.y * GBM, bn = blockIdx.x * GBN;
    int wm = (w & 1) * 64;    // 2 waves along M, each owns 64 rows
    int wn = (w >> 1) * 32;   // 4 waves along N, each owns 32 cols

    v8f acc[4][2];
#pragma unroll
    for (int i = 0; i < 4; i++)
#pragma unroll
        for (int j = 0; j < 2; j++)
#pragma unroll
            for (int e = 0; e < 8; e++) acc[i][j][e] = 0.0f;

    int lrow = tid >> 1, lhalf = tid & 1;   // 256 threads fill 128 rows x 2 half-K
    int gm = bm + lrow, gn = bn + lrow;

    for (int k0 = 0; k0 < K; k0 += GBK) {
        // --- stage A tile (f32 -> bf16, packed hw cvt) ---
        {
            BFrag pk;
            if (gm < M) {
                const float4* p = (const float4*)(X + (size_t)gm * ldx + k0 + lhalf * 16);
                pack16(pk, p[0], p[1], p[2], p[3]);
            } else {
                pk.q[0] = make_uint4(0, 0, 0, 0); pk.q[1] = make_uint4(0, 0, 0, 0);
            }
            *(uint4*)&As[lrow * LSTR + lhalf * 16]     = pk.q[0];
            *(uint4*)&As[lrow * LSTR + lhalf * 16 + 8] = pk.q[1];
        }
        // --- stage B tile (rows of W = output columns) ---
        {
            BFrag pk;
            if (gn < N) {
                const float4* p = (const float4*)(W + (size_t)gn * K + k0 + lhalf * 16);
                pack16(pk, p[0], p[1], p[2], p[3]);
            } else {
                pk.q[0] = make_uint4(0, 0, 0, 0); pk.q[1] = make_uint4(0, 0, 0, 0);
            }
            *(uint4*)&Bs[lrow * LSTR + lhalf * 16]     = pk.q[0];
            *(uint4*)&Bs[lrow * LSTR + lhalf * 16 + 8] = pk.q[1];
        }
        // prefetch next k-tile rows into cache across the barrier (global_prefetch_b8)
        if (k0 + GBK < K) {
            if (gm < M) __builtin_prefetch(X + (size_t)gm * ldx + k0 + GBK + lhalf * 16, 0, 1);
            if (gn < N) __builtin_prefetch(W + (size_t)gn * K   + k0 + GBK + lhalf * 16, 0, 1);
        }
        __syncthreads();

        // --- fragments: lane&15 = row/col, lane>>4 selects K half (CDNA5 layout) ---
        int r16 = lane & 15, kl = (lane >> 4) * 8;
        BFrag af[4], bfg[2];
#pragma unroll
        for (int mt = 0; mt < 4; mt++) {
            const unsigned short* p = &As[(wm + mt * 16 + r16) * LSTR + kl];
            af[mt].q[0] = *(const uint4*)p;
            af[mt].q[1] = *(const uint4*)(p + 16);
        }
#pragma unroll
        for (int nt = 0; nt < 2; nt++) {
            const unsigned short* p = &Bs[(wn + nt * 16 + r16) * LSTR + kl];
            bfg[nt].q[0] = *(const uint4*)p;
            bfg[nt].q[1] = *(const uint4*)(p + 16);
        }
#pragma unroll
        for (int mt = 0; mt < 4; mt++)
#pragma unroll
            for (int nt = 0; nt < 2; nt++)
                acc[mt][nt] = wmma_bf16(af[mt], bfg[nt], acc[mt][nt]);
        __syncthreads();
    }

    // --- epilogue: C layout vgpr r -> (M = r + 8*(lane>=16), N = lane&15) ---
    int c16 = lane & 15, hi = (lane >> 4) * 8;
#pragma unroll
    for (int mt = 0; mt < 4; mt++)
#pragma unroll
        for (int nt = 0; nt < 2; nt++) {
            int col = bn + wn + nt * 16 + c16;
            if (col >= N) continue;
            float bv = bias[col];
#pragma unroll
            for (int r = 0; r < 8; r++) {
                int row = bm + wm + mt * 16 + hi + r;
                if (row < M) {
                    float v = acc[mt][nt][r] + bv;
                    if (relu) v = fmaxf(v, 0.0f);
                    Y[(size_t)row * ldy + col] = v;
                }
            }
        }
}

// ---------------- embedding + positional encoding ----------------
__global__ __launch_bounds__(256)
void embed_kernel(const int* __restrict__ inputs, const float* __restrict__ tok_emb,
                  float* __restrict__ out)
{
    int row = blockIdx.x;           // b*S + s
    int b = row / SS;
    int tok = inputs[row];
    const float c = 9.210340371976184f / (float)DD;   // ln(10000)/D
    for (int j = threadIdx.x; j < DD; j += 256) {
        int i = j >> 1;
        float div = expf(-(float)(2 * i) * c);
        float ang = (float)b * div;
        float pe = (j & 1) ? cosf(ang) : sinf(ang);
        out[(size_t)row * DD + j] = tok_emb[(size_t)tok * DD + j] * 32.0f + pe;
    }
}

// ---------------- attention (per b,h,q row) ----------------
__global__ __launch_bounds__(256)
void attn_kernel(const float* __restrict__ qkv, float* __restrict__ out)
{
    __shared__ float qsh[DH];
    __shared__ float sc[SS];
    __shared__ float red[256];
    int idx = blockIdx.x;
    int q = idx % SS;
    int bh = idx / SS;
    int h = bh % NHH, b = bh / NHH;
    int tid = threadIdx.x;

    const float* qp = qkv + ((size_t)(b * SS + q)) * 3 * DD + h * DH;
    if (tid < DH) qsh[tid] = qp[tid];
    __syncthreads();

    float s;
    if (tid <= q) {
        const float* kp = qkv + ((size_t)(b * SS + tid)) * 3 * DD + DD + h * DH;
        s = 0.f;
        for (int d = 0; d < DH; d++) s += qsh[d] * kp[d];
        s *= 0.08838834764831845f;   // 1/sqrt(128)
    } else s = -INFINITY;
    sc[tid] = s; red[tid] = s;
    __syncthreads();
    for (int off = 128; off > 0; off >>= 1) {
        if (tid < off) red[tid] = fmaxf(red[tid], red[tid + off]);
        __syncthreads();
    }
    float mx = red[0];
    __syncthreads();
    float e = expf(sc[tid] - mx);
    sc[tid] = e; red[tid] = e;
    __syncthreads();
    for (int off = 128; off > 0; off >>= 1) {
        if (tid < off) red[tid] += red[tid + off];
        __syncthreads();
    }
    float inv = 1.0f / red[0];
    __syncthreads();
    if (tid < DH) {
        float o = 0.f;
        for (int k = 0; k <= q; k++)
            o += sc[k] * qkv[((size_t)(b * SS + k)) * 3 * DD + 2 * DD + h * DH + tid];
        out[((size_t)(b * SS + q)) * DD + h * DH + tid] = o * inv;
    }
}

// ---------------- y = LN(x + r) * g + b ----------------
__global__ __launch_bounds__(256)
void add_ln_kernel(const float* __restrict__ x, const float* __restrict__ r,
                   const float* __restrict__ g, const float* __restrict__ bb,
                   float* __restrict__ y)
{
    __shared__ float red[256];
    int row = blockIdx.x, tid = threadIdx.x;
    const float* xp = x + (size_t)row * DD;
    const float* rp = r + (size_t)row * DD;
    float v[4], s = 0.f;
#pragma unroll
    for (int i = 0; i < 4; i++) { v[i] = xp[tid + i * 256] + rp[tid + i * 256]; s += v[i]; }
    red[tid] = s; __syncthreads();
    for (int off = 128; off > 0; off >>= 1) { if (tid < off) red[tid] += red[tid + off]; __syncthreads(); }
    float mean = red[0] * (1.0f / DD);
    __syncthreads();
    s = 0.f;
#pragma unroll
    for (int i = 0; i < 4; i++) { float d = v[i] - mean; s += d * d; }
    red[tid] = s; __syncthreads();
    for (int off = 128; off > 0; off >>= 1) { if (tid < off) red[tid] += red[tid + off]; __syncthreads(); }
    float rstd = rsqrtf(red[0] * (1.0f / DD) + 1e-5f);
    float* yp = y + (size_t)row * DD;
#pragma unroll
    for (int i = 0; i < 4; i++) {
        int c = tid + i * 256;
        yp[c] = (v[i] - mean) * rstd * g[c] + bb[c];
    }
}

// ---------------- gate statistics ----------------
__global__ void batchwise_kernel(const float* __restrict__ gl, float* __restrict__ bw)
{
    int j = blockIdx.x * blockDim.x + threadIdx.x;
    if (j < NACTT) {
        float s = 0.f;
        for (int b = 0; b < BB; b++) s += gl[b * NACTT + j];
        bw[j] = s * (1.0f / BB);
    }
}

__global__ __launch_bounds__(256)
void stats_kernel(const float* __restrict__ bw, float* __restrict__ scal)
{
    __shared__ float r1[256], r2[256];
    int tid = threadIdx.x;
    float s = 0.f, ss = 0.f;
    for (int j = tid; j < NACTT; j += 256) { float v = bw[j]; s += v; ss += v * v; }
    r1[tid] = s; r2[tid] = ss; __syncthreads();
    for (int off = 128; off > 0; off >>= 1) {
        if (tid < off) { r1[tid] += r1[tid + off]; r2[tid] += r2[tid + off]; }
        __syncthreads();
    }
    if (tid == 0) {
        float mean = r1[0] / (float)NACTT;
        float var = (r2[0] - (float)NACTT * mean * mean) / (float)(NACTT - 1);
        float stdv = sqrtf(fmaxf(var, 0.f));
        scal[1] = mean; scal[2] = stdv;
        float cv = stdv / mean;
        scal[3] = 3.0f * cv * cv;   // importance loss
    }
}

__device__ __forceinline__ unsigned int hash_u32(unsigned int x)
{
    x ^= x >> 16; x *= 0x7feb352du; x ^= x >> 15; x *= 0x846ca68bu; x ^= x >> 16;
    return x;
}

__global__ __launch_bounds__(256)
void noise_topk_kernel(const float* __restrict__ bw, float* __restrict__ noisy,
                       const float* __restrict__ scal, float* __restrict__ out)
{
    __shared__ float bv[256];
    __shared__ int bi[256];
    int tid = threadIdx.x;
    float stdv = scal[2];
    for (int j = tid; j < NACTT; j += 256) {
        unsigned int h1 = hash_u32(2u * j + 1234u * 2654435769u);
        unsigned int h2 = hash_u32(2u * j + 1u + 1234u * 2654435769u);
        float u1 = ((h1 >> 8) + 1.0f) * (1.0f / 16777216.0f);
        float u2 = (h2 >> 8) * (1.0f / 16777216.0f);
        float gsn = sqrtf(-2.0f * logf(u1)) * cosf(6.28318530717958f * u2);
        noisy[j] = bw[j] + stdv * gsn;
    }
    __syncthreads();
    for (int it = 0; it < TOPKK; it++) {
        float best = -INFINITY; int bidx = 0x7fffffff;
        for (int j = tid; j < NACTT; j += 256) {
            float v = noisy[j];
            if (v > best || (v == best && j < bidx)) { best = v; bidx = j; }
        }
        bv[tid] = best; bi[tid] = bidx; __syncthreads();
        for (int off = 128; off > 0; off >>= 1) {
            if (tid < off) {
                if (bv[tid + off] > bv[tid] ||
                    (bv[tid + off] == bv[tid] && bi[tid + off] < bi[tid])) {
                    bv[tid] = bv[tid + off]; bi[tid] = bi[tid + off];
                }
            }
            __syncthreads();
        }
        if (tid == 0) {
            out[1 + it] = bv[0];
            out[1 + TOPKK + it] = (float)bi[0];
            noisy[bi[0]] = -INFINITY;
        }
        __syncthreads();
    }
}

// ---------------- fused decoder GEMM + online-softmax CE (stage 1) ----------------
#define CE_ASTR (DD + 8)   // 1032 ushorts, 16B aligned row stride

__global__ __launch_bounds__(256)
void ce_stage1_kernel(const float* __restrict__ h, const float* __restrict__ decw,
                      const int* __restrict__ inputs,
                      float* __restrict__ rmaxA, float* __restrict__ rsumA,
                      float* __restrict__ lablog)
{
    __shared__ unsigned short As[16 * CE_ASTR];
    __shared__ int labs[16];
    int tid = threadIdx.x, lane = tid & 31, w = tid >> 5;
    int mtile = blockIdx.y, chunk = blockIdx.x;

    if (tid < 16) {
        int tt = mtile * 16 + tid, lab = -1;
        if (tt < NCE) { int b = tt / (SS - 1), s = tt % (SS - 1); lab = inputs[b * SS + s + 1]; }
        labs[tid] = lab;
    }
    {   // stage 16 token rows of h into LDS as bf16 (full K = 1024)
        int tr = tid >> 4, c0 = (tid & 15) * 64;
        int tt = mtile * 16 + tr;
        if (tt < NCE) {
            int b = tt / (SS - 1), s = tt % (SS - 1);
            const float* hp = h + ((size_t)(b * SS + s)) * DD + c0;
            for (int j = 0; j < 64; j += 16) {
                const float4* p = (const float4*)(hp + j);
                BFrag pk;
                pack16(pk, p[0], p[1], p[2], p[3]);
                *(uint4*)&As[tr * CE_ASTR + c0 + j]     = pk.q[0];
                *(uint4*)&As[tr * CE_ASTR + c0 + j + 8] = pk.q[1];
            }
        } else {
            for (int j = 0; j < 64; j++) As[tr * CE_ASTR + c0 + j] = 0;
        }
    }
    __syncthreads();

    int tiles_per_chunk = (NTILE_V + CE_CHUNKS - 1) / CE_CHUNKS;
    int t0 = chunk * tiles_per_chunk;
    int t1 = min(NTILE_V, t0 + tiles_per_chunk);

    float rmax[8], rsum[8];
#pragma unroll
    for (int r = 0; r < 8; r++) { rmax[r] = -INFINITY; rsum[r] = 0.f; }

    int r16 = lane & 15, kl = (lane >> 4) * 8;

    for (int tile = t0 + w; tile < t1; tile += 8) {
        int n0 = tile * 16;
        int n = n0 + r16;
        bool cv = n < VV;
        v8f acc;
#pragma unroll
        for (int e = 0; e < 8; e++) acc[e] = 0.f;

        for (int k0 = 0; k0 < DD; k0 += 32) {
            BFrag af, bfg;
            const unsigned short* ap = &As[r16 * CE_ASTR + k0 + kl];
            af.q[0] = *(const uint4*)ap;
            af.q[1] = *(const uint4*)(ap + 16);
            if (cv) {
                const float* wp = decw + (size_t)n * DD + k0 + kl;
                float4 f0 = *(const float4*)wp;
                float4 f1 = *(const float4*)(wp + 4);
                float4 f2 = *(const float4*)(wp + 16);
                float4 f3 = *(const float4*)(wp + 20);
                pack16(bfg, f0, f1, f2, f3);
            } else {
                bfg.q[0] = make_uint4(0, 0, 0, 0); bfg.q[1] = make_uint4(0, 0, 0, 0);
            }
            acc = wmma_bf16(af, bfg, acc);
        }

#pragma unroll
        for (int r = 0; r < 8; r++) {
            float x = cv ? acc[r] : -INFINITY;
            float orig = x;
            float tmax = x;
            tmax = fmaxf(tmax, __shfl_xor(tmax, 1, 16));
            tmax = fmaxf(tmax, __shfl_xor(tmax, 2, 16));
            tmax = fmaxf(tmax, __shfl_xor(tmax, 4, 16));
            tmax = fmaxf(tmax, __shfl_xor(tmax, 8, 16));
            float e = expf(x - tmax);
            float tsum = e;
            tsum += __shfl_xor(tsum, 1, 16);
            tsum += __shfl_xor(tsum, 2, 16);
            tsum += __shfl_xor(tsum, 4, 16);
            tsum += __shfl_xor(tsum, 8, 16);
            float nm = fmaxf(rmax[r], tmax);
            rsum[r] = rsum[r] * expf(rmax[r] - nm) + tsum * expf(tmax - nm);
            rmax[r] = nm;
            int m = r + ((lane >= 16) ? 8 : 0);
            if (cv && labs[m] == n) lablog[mtile * 16 + m] = orig;
        }
    }

    int slot = chunk * 8 + w;
    if (lane == 0) {
#pragma unroll
        for (int r = 0; r < 8; r++) {
            rmaxA[(size_t)slot * NTOK + mtile * 16 + r] = rmax[r];
            rsumA[(size_t)slot * NTOK + mtile * 16 + r] = rsum[r];
        }
    }
    if (lane == 16) {
#pragma unroll
        for (int r = 0; r < 8; r++) {
            rmaxA[(size_t)slot * NTOK + mtile * 16 + 8 + r] = rmax[r];
            rsumA[(size_t)slot * NTOK + mtile * 16 + 8 + r] = rsum[r];
        }
    }
}

// ---------------- CE stage 2: merge partials, accumulate -logp ----------------
__global__ void ce_stage2_kernel(const float* __restrict__ rmaxA, const float* __restrict__ rsumA,
                                 const float* __restrict__ lablog, float* __restrict__ scal)
{
    int t = blockIdx.x * blockDim.x + threadIdx.x;
    if (t >= NCE) return;
    float M = -INFINITY;
    for (int s = 0; s < CE_CHUNKS * 8; s++) M = fmaxf(M, rmaxA[(size_t)s * NTOK + t]);
    float Z = 0.f;
    for (int s = 0; s < CE_CHUNKS * 8; s++)
        Z += rsumA[(size_t)s * NTOK + t] * expf(rmaxA[(size_t)s * NTOK + t] - M);
    float lp = lablog[t] - M - logf(Z);
    atomicAdd(&scal[0], -lp);
}

__global__ void init_kernel(float* scal) { if (threadIdx.x < 16) scal[threadIdx.x] = 0.f; }

__global__ void final_kernel(const float* __restrict__ scal, float* __restrict__ out)
{
    out[0] = scal[0] * (1.0f / (float)NCE) + scal[3];
}

// ---------------- host-side layer driver ----------------
static void run_layer(float* x, float* t1, float* t2, float* qkv, float* ffh,
                      const float* qkv_w, const float* qkv_b,
                      const float* out_w, const float* out_b,
                      const float* l1g, const float* l1b,
                      const float* l2g, const float* l2b,
                      const float* f1w, const float* f1b,
                      const float* f2w, const float* f2b,
                      hipStream_t stream)
{
    gemm_xwT<<<dim3(3 * DD / GBN, NTOK / GBM), 256, 0, stream>>>(
        x, DD, qkv_w, qkv_b, qkv, 3 * DD, NTOK, 3 * DD, DD, 0);
    attn_kernel<<<BB * NHH * SS, 256, 0, stream>>>(qkv, t1);
    gemm_xwT<<<dim3(DD / GBN, NTOK / GBM), 256, 0, stream>>>(
        t1, DD, out_w, out_b, t2, DD, NTOK, DD, DD, 0);
    add_ln_kernel<<<NTOK, 256, 0, stream>>>(x, t2, l1g, l1b, x);
    gemm_xwT<<<dim3(HIDD / GBN, NTOK / GBM), 256, 0, stream>>>(
        x, DD, f1w, f1b, ffh, HIDD, NTOK, HIDD, DD, 1);
    gemm_xwT<<<dim3(DD / GBN, NTOK / GBM), 256, 0, stream>>>(
        ffh, HIDD, f2w, f2b, t2, DD, NTOK, DD, HIDD, 0);
    add_ln_kernel<<<NTOK, 256, 0, stream>>>(x, t2, l2g, l2b, x);
}

extern "C" void kernel_launch(void* const* d_in, const int* in_sizes, int n_in,
                              void* d_out, int out_size, void* d_ws, size_t ws_size,
                              hipStream_t stream)
{
    (void)in_sizes; (void)n_in; (void)out_size; (void)ws_size;
    const int*   inputs  = (const int*)d_in[0];
    const float* resp    = (const float*)d_in[1];
    const float* tok_emb = (const float*)d_in[2];
    const float* gate_w  = (const float*)d_in[3];
    const float* gate_b  = (const float*)d_in[4];
    const float* r_qkv_w = (const float*)d_in[5];
    const float* r_qkv_b = (const float*)d_in[6];
    const float* r_out_w = (const float*)d_in[7];
    const float* r_out_b = (const float*)d_in[8];
    const float* r_ln1_g = (const float*)d_in[9];
    const float* r_ln1_b = (const float*)d_in[10];
    const float* r_ln2_g = (const float*)d_in[11];
    const float* r_ln2_b = (const float*)d_in[12];
    const float* r_ff1_w = (const float*)d_in[13];
    const float* r_ff1_b = (const float*)d_in[14];
    const float* r_ff2_w = (const float*)d_in[15];
    const float* r_ff2_b = (const float*)d_in[16];
    const float* e_qkv_w = (const float*)d_in[17];
    const float* e_qkv_b = (const float*)d_in[18];
    const float* e_out_w = (const float*)d_in[19];
    const float* e_out_b = (const float*)d_in[20];
    const float* e_ln1_g = (const float*)d_in[21];
    const float* e_ln1_b = (const float*)d_in[22];
    const float* e_ln2_g = (const float*)d_in[23];
    const float* e_ln2_b = (const float*)d_in[24];
    const float* e_ff1_w = (const float*)d_in[25];
    const float* e_ff1_b = (const float*)d_in[26];
    const float* e_ff2_w = (const float*)d_in[27];
    const float* e_ff2_b = (const float*)d_in[28];
    const float* dec_w   = (const float*)d_in[29];
    float* out = (float*)d_out;

    // workspace layout (floats)
    float* ws = (float*)d_ws;
    size_t off = 0;
    float* xbuf  = ws + off; off += (size_t)NTOK * DD;
    float* t1    = ws + off; off += (size_t)NTOK * DD;
    float* t2    = ws + off; off += (size_t)NTOK * DD;
    float* qkv   = ws + off; off += (size_t)NTOK * 3 * DD;
    float* ffh   = ws + off; off += (size_t)NTOK * HIDD;
    float* gl    = ws + off; off += (size_t)BB * NACTT;
    float* bw    = ws + off; off += NACTT;
    float* noisy = ws + off; off += NACTT;
    float* scal  = ws + off; off += 16;
    float* rmaxA = ws + off; off += (size_t)CE_CHUNKS * 8 * NTOK;
    float* rsumA = ws + off; off += (size_t)CE_CHUNKS * 8 * NTOK;
    float* lablog= ws + off; off += NTOK;

    init_kernel<<<1, 32, 0, stream>>>(scal);

    // routing path
    embed_kernel<<<NTOK, 256, 0, stream>>>(inputs, tok_emb, xbuf);
    run_layer(xbuf, t1, t2, qkv, ffh,
              r_qkv_w, r_qkv_b, r_out_w, r_out_b,
              r_ln1_g, r_ln1_b, r_ln2_g, r_ln2_b,
              r_ff1_w, r_ff1_b, r_ff2_w, r_ff2_b, stream);
    gemm_xwT<<<dim3((NACTT + GBN - 1) / GBN, 1), 256, 0, stream>>>(
        xbuf + (size_t)(SS - 1) * DD, (long)SS * DD,
        gate_w, gate_b, gl, NACTT, BB, NACTT, DD, 0);
    batchwise_kernel<<<(NACTT + 255) / 256, 256, 0, stream>>>(gl, bw);
    stats_kernel<<<1, 256, 0, stream>>>(bw, scal);
    noise_topk_kernel<<<1, 256, 0, stream>>>(bw, noisy, scal, out);

    // expert path
    hipMemcpyAsync(xbuf, resp, (size_t)NTOK * DD * sizeof(float),
                   hipMemcpyDeviceToDevice, stream);
    for (int i = 0; i < LLYR; i++) {
        run_layer(xbuf, t1, t2, qkv, ffh,
                  e_qkv_w + (size_t)i * 3 * DD * DD, e_qkv_b + (size_t)i * 3 * DD,
                  e_out_w + (size_t)i * DD * DD,     e_out_b + (size_t)i * DD,
                  e_ln1_g + (size_t)i * DD, e_ln1_b + (size_t)i * DD,
                  e_ln2_g + (size_t)i * DD, e_ln2_b + (size_t)i * DD,
                  e_ff1_w + (size_t)i * HIDD * DD, e_ff1_b + (size_t)i * HIDD,
                  e_ff2_w + (size_t)i * DD * HIDD, e_ff2_b + (size_t)i * DD, stream);
    }

    // fused decoder + CE
    ce_stage1_kernel<<<dim3(CE_CHUNKS, NTOK / 16), 256, 0, stream>>>(
        xbuf, dec_w, inputs, rmaxA, rsumA, lablog);
    ce_stage2_kernel<<<(NCE + 255) / 256, 256, 0, stream>>>(rmaxA, rsumA, lablog, scal);
    final_kernel<<<1, 1, 0, stream>>>(scal, out);
}
